// AtomConv_17437567222207
// MI455X (gfx1250) — compile-verified, matrix operand order checked by reading, not run on hardware
//
#include <hip/hip_runtime.h>
#include <hip/hip_bf16.h>

// ---------------- problem constants (match reference) ----------------
#define N_ATOMS   100000
#define N_DIR     1600000
#define N_UND     800000
#define ATOM_DIM  64
#define HIDDEN    64
#define IN_DIM    192      // 2*ATOM_DIM + BOND_DIM

// ---------------- WMMA helper types ----------------
typedef __attribute__((ext_vector_type(16))) __bf16 v16bf;
typedef __attribute__((ext_vector_type(8)))  float  v8f;

struct U16x16 { unsigned short u[16]; };   // 32B, for bit_cast -> v16bf
struct Frag32B { uint4 lo, hi; };          // 32B LDS fragment load

static __device__ inline unsigned short f2bf(float f) {
  // round-to-nearest-even f32 -> bf16
  unsigned int u = __builtin_bit_cast(unsigned int, f);
  u += 0x7FFFu + ((u >> 16) & 1u);
  return (unsigned short)(u >> 16);
}

static __device__ inline unsigned int pack2bf(float lo, float hi) {
  return (unsigned int)f2bf(lo) | ((unsigned int)f2bf(hi) << 16);
}

static __device__ inline float sigm(float x) { return 1.0f / (1.0f + __expf(-x)); }
static __device__ inline float silu(float x) { return x * sigm(x); }

// k_local (0..31) -> (lane-half bit, element index i in v16bf) for 16-bit frags
static __device__ inline void sw_pos(int kl, int& hf, int& ii) {
  int within = kl & 15;
  hf = (within >> 3) & 1;
  ii = ((kl >> 4) << 3) + (within & 7);
}

static __device__ inline v16bf ld_frag(const unsigned short* p16 /*16B aligned*/) {
  const uint4* p = (const uint4*)p16;
  Frag32B f; f.lo = p[0]; f.hi = p[1];
  return __builtin_bit_cast(v16bf, f);
}

static __device__ inline v8f wmma_bf16(v16bf a, v16bf b, v8f c) {
  return __builtin_amdgcn_wmma_f32_16x16x32_bf16(false, a, false, b, (short)0, c, false, false);
}

static __device__ inline v8f bcast8(float x) {
  v8f r;
  #pragma unroll
  for (int j = 0; j < 8; ++j) r[j] = x;
  return r;
}

// ---------------- kernel 0: zero the atom accumulator ----------------
__global__ void zero_kernel(float* __restrict__ p, int n) {
  int i = blockIdx.x * blockDim.x + threadIdx.x;
  if (i < n) p[i] = 0.0f;
}

// ---------------- kernel 1: per-edge gated MLP + atomic scatter ----------------
// Block = 256 threads (8 waves). Each wave processes 16-edge tiles.
__global__ __launch_bounds__(256) void edge_mlp_kernel(
    const float* __restrict__ atom_feas,   // [N_ATOMS,64]
    const float* __restrict__ bond_feas,   // [N_UND,64]
    const float* __restrict__ bond_w,      // [N_UND,64]
    const int*   __restrict__ atom_graph,  // [N_DIR,2]
    const int*   __restrict__ d2u,         // [N_DIR]
    const float* __restrict__ W1c, const float* __restrict__ b1c,
    const float* __restrict__ W2c, const float* __restrict__ b2c,
    const float* __restrict__ W1g, const float* __restrict__ b1g,
    const float* __restrict__ W2g, const float* __restrict__ b2g,
    float* __restrict__ acc_out)           // [N_ATOMS,64] f32 accumulator (ws)
{
  // Pre-swizzled bf16 fragment stores; 512 ushorts per 32x16 fragment tile.
  __shared__ alignas(16) unsigned short sW1[2 * 6 * 4 * 512];   // 48 KB {c,g} x kb x nt
  __shared__ alignas(16) unsigned short sW2[2 * 2 * 4 * 512];   // 16 KB
  __shared__ alignas(16) unsigned short sMsg[8 * 6 * 512];      // 48 KB per-wave A frags
  __shared__ alignas(16) unsigned short sHid[8 * 2 * 2 * 512];  // 32 KB per-wave hidden A frags

  const int tid  = threadIdx.x;
  const int lane = tid & 31;
  const int wave = tid >> 5;
  const int half = lane >> 4;
  const int l16  = lane & 15;

  // ---- stage layer-1 weights (both paths), swizzled to B-fragment layout ----
  for (int idx = tid; idx < 2 * IN_DIM * HIDDEN; idx += 256) {
    int mat = idx / (IN_DIM * HIDDEN);
    int rem = idx - mat * (IN_DIM * HIDDEN);
    int k = rem >> 6, n = rem & 63;
    float v = (mat ? W1g : W1c)[rem];
    int kb = k >> 5, hf, ii; sw_pos(k & 31, hf, ii);
    sW1[(((mat * 6 + kb) << 2) + (n >> 4)) * 512 + ((n & 15) + (hf << 4)) * 16 + ii] = f2bf(v);
  }
  // ---- stage layer-2 weights ----
  for (int idx = tid; idx < 2 * HIDDEN * ATOM_DIM; idx += 256) {
    int mat = idx >> 12;
    int rem = idx & 4095;
    int k = rem >> 6, n = rem & 63;
    float v = (mat ? W2g : W2c)[rem];
    int kb = k >> 5, hf, ii; sw_pos(k & 31, hf, ii);
    sW2[(((mat * 2 + kb) << 2) + (n >> 4)) * 512 + ((n & 15) + (hf << 4)) * 16 + ii] = f2bf(v);
  }
  __syncthreads();

  unsigned short* msgW = &sMsg[wave * 6 * 512];
  unsigned short* hidW = &sHid[wave * 4 * 512];

  // Loop-invariant staging geometry: lane owns even column pair (c0, c0+1) of
  // each 64-wide msg segment; even K pairs share (hf, ii/ii+1) -> one b32 store.
  const int c0 = lane << 1;                 // 0..62
  int hf0, ii0; sw_pos(c0 & 31, hf0, ii0);
  const int base0 = (((c0 >> 5) * 32 + (hf0 << 4)) << 4) + ii0;
  const int k1 = 64 + c0;
  int hf1, ii1; sw_pos(k1 & 31, hf1, ii1);
  const int base1 = (((k1 >> 5) * 32 + (hf1 << 4)) << 4) + ii1;
  const int k2 = 128 + c0;
  int hf2, ii2; sw_pos(k2 & 31, hf2, ii2);
  const int base2 = (((k2 >> 5) * 32 + (hf2 << 4)) << 4) + ii2;

  const int nTiles = N_DIR / 16;
  for (int tile = blockIdx.x * 8 + wave; tile < nTiles; tile += gridDim.x * 8) {
    const int eBase = tile << 4;

    // Per-lane edge indices for row (lane & 15); broadcast via shuffles below.
    const int eRow = eBase + l16;
    const int cenL = atom_graph[2 * eRow];
    const int nbrL = atom_graph[2 * eRow + 1];
    const int undL = d2u[eRow];

    // ---- gather 16 msg rows (192 f32) -> bf16 A-fragment swizzle in LDS ----
    #pragma unroll 1
    for (int r = 0; r < 16; ++r) {
      const int center = __shfl(cenL, r, 32);
      const int nbr    = __shfl(nbrL, r, 32);
      const int und    = __shfl(undL, r, 32);
      float2 va = *(const float2*)&atom_feas[center * 64 + c0];
      float2 vb = *(const float2*)&bond_feas[und    * 64 + c0];
      float2 vc = *(const float2*)&atom_feas[nbr    * 64 + c0];
      const int roff = r << 4;
      *(unsigned int*)&msgW[base0 + roff] = pack2bf(va.x, va.y);
      *(unsigned int*)&msgW[base1 + roff] = pack2bf(vb.x, vb.y);
      *(unsigned int*)&msgW[base2 + roff] = pack2bf(vc.x, vc.y);
    }

    // ---- layer 1: [16x192] @ [192x64] x2 paths, one 16-col tile at a time ----
    #pragma unroll 1
    for (int nt = 0; nt < 4; ++nt) {
      v8f aC = bcast8(b1c[(nt << 4) + l16]);
      v8f aG = bcast8(b1g[(nt << 4) + l16]);
      #pragma unroll
      for (int kb = 0; kb < 6; ++kb) {
        v16bf A  = ld_frag(&msgW[(kb * 32 + lane) * 16]);
        v16bf Bc = ld_frag(&sW1[((kb << 2) + nt) * 512 + lane * 16]);
        aC = wmma_bf16(A, Bc, aC);
        v16bf Bg = ld_frag(&sW1[(((6 + kb) << 2) + nt) * 512 + lane * 16]);
        aG = wmma_bf16(A, Bg, aG);
      }
      // SiLU + re-swizzle hidden (C/D layout -> A-fragment layout) via LDS
      #pragma unroll
      for (int j = 0; j < 8; ++j) {
        int m = j + (half << 3);            // row within tile
        int n = l16 + (nt << 4);            // hidden index -> layer-2 K
        int kb2 = n >> 5, hf, ii; sw_pos(n & 31, hf, ii);
        int ln2 = m + (hf << 4);
        hidW[((0 + kb2) * 32 + ln2) * 16 + ii] = f2bf(silu(aC[j]));
        hidW[((2 + kb2) * 32 + ln2) * 16 + ii] = f2bf(silu(aG[j]));
      }
    }

    // ---- layer 2 + activation + gather bond_w + atomic scatter, per nt ----
    #pragma unroll 1
    for (int nt = 0; nt < 4; ++nt) {
      v8f c2 = bcast8(b2c[(nt << 4) + l16]);
      v8f g2 = bcast8(b2g[(nt << 4) + l16]);
      #pragma unroll
      for (int kb = 0; kb < 2; ++kb) {
        v16bf Ac = ld_frag(&hidW[((0 + kb) * 32 + lane) * 16]);
        v16bf Bc = ld_frag(&sW2[((kb << 2) + nt) * 512 + lane * 16]);
        c2 = wmma_bf16(Ac, Bc, c2);
        v16bf Ag = ld_frag(&hidW[((2 + kb) * 32 + lane) * 16]);
        v16bf Bg = ld_frag(&sW2[(((2 + kb) << 2) + nt) * 512 + lane * 16]);
        g2 = wmma_bf16(Ag, Bg, g2);
      }
      const int n = l16 + (nt << 4);
      #pragma unroll
      for (int j = 0; j < 8; ++j) {
        int m = j + (half << 3);
        int und    = __shfl(undL, m, 32);
        int center = __shfl(cenL, m, 32);
        float val = silu(c2[j]) * sigm(g2[j]) * bond_w[und * 64 + n];
        unsafeAtomicAdd(&acc_out[center * 64 + n], val);  // global_atomic_add_f32
      }
    }
  }
}

// ---------------- kernel 2: out = acc @ Wo + bo + atom_feas ----------------
__global__ __launch_bounds__(256) void out_proj_kernel(
    const float* __restrict__ accum,     // [N_ATOMS,64] f32
    const float* __restrict__ atom_feas, // [N_ATOMS,64]
    const float* __restrict__ Wo,        // [64,64]
    const float* __restrict__ bo,        // [64]
    float* __restrict__ out)             // [N_ATOMS,64]
{
  __shared__ alignas(16) unsigned short sWo[2 * 4 * 512];  // 8 KB swizzled B frags

  const int tid  = threadIdx.x;
  const int lane = tid & 31;
  const int wave = tid >> 5;
  const int half = lane >> 4;
  const int l16  = lane & 15;

  for (int idx = tid; idx < 64 * 64; idx += 256) {
    int k = idx >> 6, n = idx & 63;
    int kb = k >> 5, hf, ii; sw_pos(k & 31, hf, ii);
    sWo[((kb << 2) + (n >> 4)) * 512 + ((n & 15) + (hf << 4)) * 16 + ii] = f2bf(Wo[idx]);
  }
  __syncthreads();

  const int nTiles = N_ATOMS / 16;  // 6250
  #pragma unroll 1
  for (int tile = blockIdx.x * 8 + wave; tile < nTiles; tile += gridDim.x * 8) {
    const int aBase = tile << 4;

    // Build both bf16 A fragments (K=0..63) from the f32 accumulator (row = l16).
    v16bf Afrag[2];
    #pragma unroll
    for (int kb = 0; kb < 2; ++kb) {
      const float* row = &accum[(aBase + l16) * 64 + kb * 32 + (half << 3)];
      float4 f0 = *(const float4*)&row[0];
      float4 f1 = *(const float4*)&row[4];
      float4 f2 = *(const float4*)&row[16];
      float4 f3 = *(const float4*)&row[20];
      float fv[16] = { f0.x, f0.y, f0.z, f0.w, f1.x, f1.y, f1.z, f1.w,
                       f2.x, f2.y, f2.z, f2.w, f3.x, f3.y, f3.z, f3.w };
      U16x16 t;
      #pragma unroll
      for (int i = 0; i < 16; ++i) t.u[i] = f2bf(fv[i]);
      Afrag[kb] = __builtin_bit_cast(v16bf, t);
    }

    #pragma unroll 1
    for (int nt = 0; nt < 4; ++nt) {
      // bias + residual init in C/D layout
      v8f acc = bcast8(bo[(nt << 4) + l16]);
      #pragma unroll
      for (int j = 0; j < 8; ++j) {
        int m = j + (half << 3);
        acc[j] += atom_feas[(aBase + m) * 64 + l16 + (nt << 4)];
      }
      #pragma unroll
      for (int kb = 0; kb < 2; ++kb) {
        v16bf B = ld_frag(&sWo[((kb << 2) + nt) * 512 + lane * 16]);
        acc = wmma_bf16(Afrag[kb], B, acc);
      }
      #pragma unroll
      for (int j = 0; j < 8; ++j) {
        int m = j + (half << 3);
        out[(aBase + m) * 64 + l16 + (nt << 4)] = acc[j];
      }
    }
  }
}

// ---------------- host launcher ----------------
extern "C" void kernel_launch(void* const* d_in, const int* in_sizes, int n_in,
                              void* d_out, int out_size, void* d_ws, size_t ws_size,
                              hipStream_t stream) {
  const float* atom_feas = (const float*)d_in[0];
  const float* bond_feas = (const float*)d_in[1];
  const float* bond_w    = (const float*)d_in[2];
  const int*   atom_graph= (const int*)  d_in[3];
  const int*   d2u       = (const int*)  d_in[4];
  const float* W1c = (const float*)d_in[5];
  const float* b1c = (const float*)d_in[6];
  const float* W2c = (const float*)d_in[7];
  const float* b2c = (const float*)d_in[8];
  const float* W1g = (const float*)d_in[9];
  const float* b1g = (const float*)d_in[10];
  const float* W2g = (const float*)d_in[11];
  const float* b2g = (const float*)d_in[12];
  const float* Wo  = (const float*)d_in[13];
  const float* bo  = (const float*)d_in[14];

  float* acc = (float*)d_ws;                       // N_ATOMS*64 f32 accumulator
  (void)ws_size; (void)n_in; (void)in_sizes; (void)out_size;

  const int accN = N_ATOMS * ATOM_DIM;
  zero_kernel<<<(accN + 255) / 256, 256, 0, stream>>>(acc, accN);

  edge_mlp_kernel<<<2048, 256, 0, stream>>>(
      atom_feas, bond_feas, bond_w, atom_graph, d2u,
      W1c, b1c, W2c, b2c, W1g, b1g, W2g, b2g, acc);

  out_proj_kernel<<<512, 256, 0, stream>>>(acc, atom_feas, Wo, bo, (float*)d_out);
}
